// UnifiedInteractionLayer_88776974008838
// MI455X (gfx1250) — matrix operand order repaired and checked by reading, not compiled, not run on hardware
//
#include <hip/hip_runtime.h>
#include <hip/hip_bf16.h>

typedef __attribute__((ext_vector_type(16))) _Float16 v16h;
typedef __attribute__((ext_vector_type(8)))  float    v8f;

#define S_DIM   64
#define V_DIM   32
#define L2_DIM  16
#define T_DIM   64
#define D_DIM   336     // S + 6V + 5*L2
#define ESD     252
#define ESD_PAD 256
#define HID     128
#define WDIM    432     // 27 * 16 exactly
#define MSG_DIM 336
#define TE      16      // edges per block (one WMMA M tile)

__device__ __forceinline__ float silu_f(float x) { return x / (1.0f + __expf(-x)); }

// ---------------------------------------------------------------------------
// prep: zero agg accumulator, build f16 weight copies in B-fragment order.
// B fragment (16x16x32 f16 wmma): lane L holds col n = L&15; elements e=0..15
// hold K rows k = (e>>3)*16 + (L>>4)*8 + (e&7) of the 32-deep K slice. One tile = 32 lanes *
// 16 halfs = 1024B contiguous, so each lane loads its fragment as 32
// contiguous bytes.
// ---------------------------------------------------------------------------
__global__ void prep_kernel(const float* __restrict__ W1, const float* __restrict__ W2,
                            _Float16* __restrict__ W1sw, _Float16* __restrict__ W2sw,
                            float* __restrict__ agg, long aggN)
{
    const long stride = (long)gridDim.x * blockDim.x;
    const long idx0   = (long)blockIdx.x * blockDim.x + threadIdx.x;
    for (long i = idx0; i < aggN; i += stride) agg[i] = 0.0f;
    // W1sw: 8 nTiles x 8 kChunks x 512 halfs
    for (long i = idx0; i < (long)8 * 8 * 512; i += stride) {
        int t = (int)(i >> 9), w = (int)(i & 511);
        int lane = w >> 4, e = w & 15;
        int nT = t >> 3, kc = t & 7;
        int k = kc * 32 + ((e >> 3) << 4) + (lane >> 4) * 8 + (e & 7);
        int n = nT * 16 + (lane & 15);
        W1sw[i] = (k < ESD) ? (_Float16)W1[k * HID + n] : (_Float16)0.0f;
    }
    // W2sw: 27 nTiles x 4 kChunks x 512 halfs
    for (long i = idx0; i < (long)27 * 4 * 512; i += stride) {
        int t = (int)(i >> 9), w = (int)(i & 511);
        int lane = w >> 4, e = w & 15;
        int nT = t >> 2, kc = t & 3;
        int k = kc * 32 + ((e >> 3) << 4) + (lane >> 4) * 8 + (e & 7);
        int n = nT * 16 + (lane & 15);
        W2sw[i] = (_Float16)W2[k * WDIM + n];
    }
}

// ---------------------------------------------------------------------------
// Fused edge kernel: feature build -> WMMA MLP -> message build -> atomic agg
// ---------------------------------------------------------------------------
__global__ __launch_bounds__(128)
void edge_kernel(const float* __restrict__ h, const float* __restrict__ coords,
                 const int* __restrict__ ei,
                 const int* __restrict__ etype, const int* __restrict__ ebt,
                 const int* __restrict__ ebc, const int* __restrict__ ebr,
                 const int* __restrict__ ebs, const float* __restrict__ erd,
                 const float* __restrict__ temb,
                 const float* __restrict__ etype_emb, const float* __restrict__ bte,
                 const float* __restrict__ bce, const float* __restrict__ bre,
                 const float* __restrict__ bse,
                 const float* __restrict__ refW, const float* __restrict__ refb,
                 const float* __restrict__ b1, const float* __restrict__ b2,
                 const float* __restrict__ Wvs, const float* __restrict__ Wls,
                 const float* __restrict__ Wsv, const float* __restrict__ Wlv,
                 const float* __restrict__ Wvl, const float* __restrict__ Wsl,
                 const _Float16* __restrict__ W1sw, const _Float16* __restrict__ W2sw,
                 float* __restrict__ agg, int E)
{
    // WMMA staging (A fragments) overlaid with phase-4 scratch (dead by then)
    __shared__ alignas(32) union {
        struct { _Float16 esA[8 * 32 * 16]; _Float16 hidA[4 * 32 * 16]; } g; // 12288 B
        struct { float t112[TE * V_DIM * 5]; float d1[TE * V_DIM]; } p;      // 12288 B
    } u;
    __shared__ alignas(32) _Float16 wbuf[TE * WDIM];  // MLP output w, f16
    __shared__ float sHsrc[TE * D_DIM];
    __shared__ float sHsv[TE * V_DIM];
    __shared__ float sD2[TE * L2_DIM];
    __shared__ float sHsl[TE * L2_DIM];
    __shared__ float sQ[TE * L2_DIM * 3];
    __shared__ float sY1[TE * 3];
    __shared__ float sY2[TE * 5];
    __shared__ float sBy[TE * 5 * 3];
    __shared__ float sDist[TE];
    __shared__ int   sSrc[TE], sDst[TE];

    const int tid = threadIdx.x;
    const int e0  = blockIdx.x * TE;

    // ---- phase 0: gather h[src] rows + per-edge geometry ----
    for (int idx = tid; idx < TE * D_DIM; idx += 128) {
        int m = idx / D_DIM, c = idx - m * D_DIM;
        int e = (e0 + m < E) ? (e0 + m) : (E - 1);
        int s = ei[e];
        sHsrc[m * D_DIM + c] = h[(long)s * D_DIM + c];
    }
    if (tid < TE) {
        int e = (e0 + tid < E) ? (e0 + tid) : (E - 1);
        int s = ei[e], d = ei[(long)E + e];
        sSrc[tid] = s; sDst[tid] = d;
        float dx = coords[d * 3 + 0] - coords[s * 3 + 0];
        float dy = coords[d * 3 + 1] - coords[s * 3 + 1];
        float dz = coords[d * 3 + 2] - coords[s * 3 + 2];
        float dist = sqrtf(dx * dx + dy * dy + dz * dz + 1e-12f);
        sDist[tid] = dist;
        float x = dx / dist, y = dy / dist, z = dz / dist;
        sY1[tid * 3 + 0] = x; sY1[tid * 3 + 1] = y; sY1[tid * 3 + 2] = z;
        const float r2 = 0.7071067811865475f, r6 = 0.4082482904638631f;
        sY2[tid * 5 + 0] = 2.f * r2 * x * y;
        sY2[tid * 5 + 1] = 2.f * r2 * x * z;
        sY2[tid * 5 + 2] = 2.f * r2 * y * z;
        sY2[tid * 5 + 3] = r2 * (x * x - y * y);
        sY2[tid * 5 + 4] = r6 * (x * x + y * y - 2.f * z * z);
        float* By = &sBy[tid * 15];
        By[0]  = r2 * y; By[1]  = r2 * x;  By[2]  = 0.f;
        By[3]  = r2 * z; By[4]  = 0.f;     By[5]  = r2 * x;
        By[6]  = 0.f;    By[7]  = r2 * z;  By[8]  = r2 * y;
        By[9]  = r2 * x; By[10] = -r2 * y; By[11] = 0.f;
        By[12] = r6 * x; By[13] = r6 * y;  By[14] = -2.f * r6 * z;
    }
    __syncthreads();

    // ---- phase 1: build es directly in 16-bit A-fragment layout ----
    for (int idx = tid; idx < TE * ESD_PAD; idx += 128) {
        int m = idx >> 8, k = idx & 255;
        int e = (e0 + m < E) ? (e0 + m) : (E - 1);
        float v = 0.0f;
        if (k < 16) {                       // RBF: exp(-(3d-k)^2), CUTOFF=5, NRBF=16
            float t = 3.0f * sDist[m] - (float)k;
            v = __expf(-t * t);
        } else if (k < 32) {                // edge type embedding
            v = etype_emb[etype[e] * 16 + (k - 16)];
        } else if (k < 52) {                // bond embeddings
            int j = k - 32;
            int bt = ebt[e];
            bool isb = bt >= 0;
            if (j < 8)       v = bte[(bt + 1) * 8 + j];
            else if (j < 12) v = bce[(isb ? ebc[e] : 2) * 4 + (j - 8)];
            else if (j < 16) v = bre[(isb ? ebr[e] : 2) * 4 + (j - 12)];
            else             v = bse[(ebs[e] + 1) * 4 + (j - 16)];
        } else if (k < 60) {                // e_ref = [|dd|, dd, has_ref] @ refW + refb
            int j = k - 52;
            float rd = erd[e];
            float dd = sDist[m] - rd;
            float hr = rd > 0.0f ? 1.0f : 0.0f;
            v = fabsf(dd) * refW[j] + dd * refW[8 + j] + hr * refW[16 + j] + refb[j];
        } else if (k < 124) {               // h0[src]
            v = sHsrc[m * D_DIM + (k - 60)];
        } else if (k < 188) {               // h0[dst]
            v = h[(long)sDst[m] * D_DIM + (k - 124)];
        } else if (k < 252) {               // t_emb[dst]
            v = temb[(long)sDst[m] * T_DIM + (k - 188)];
        }
        int kk = k & 31, chk = k >> 5;
        int lane2 = ((kk >> 3) & 1) * 16 + m;
        int e2 = ((kk >> 4) << 3) | (kk & 7);
        u.g.esA[(chk * 32 + lane2) * 16 + e2] = (_Float16)v;
    }
    __syncthreads();

    const int lane = tid & 31;
    const int wv   = tid >> 5;
    const int nlow = lane & 15;
    const int gg   = lane >> 4;

    // ---- phase 2: GEMM1  (16 x 128) = es(16x256) @ W1  + b1, silu -> hidA ----
    // A fragments loaded ONCE into registers, reused across both N tiles.
    {
        v16h aF[8];
#pragma unroll
        for (int kc = 0; kc < 8; ++kc)
            aF[kc] = *(const v16h*)&u.g.esA[(kc * 32 + lane) * 16];
        const v16h* Bp = (const v16h*)W1sw;
#pragma unroll
        for (int t = 0; t < 2; ++t) {
            int nT = wv * 2 + t;
            v8f acc = {};
#pragma unroll
            for (int kc = 0; kc < 8; ++kc) {
                v16h b = Bp[(nT * 8 + kc) * 32 + lane];
                acc = __builtin_amdgcn_wmma_f32_16x16x32_f16(
                        false, aF[kc], false, b, (short)0, acc, false, false);
            }
            int n = nT * 16 + nlow;
            float bias = b1[n];
            int kk = n & 31, chk = n >> 5;
            int lbase = ((kk >> 3) & 1) * 16;
            int e2 = ((kk >> 4) << 3) | (kk & 7);
#pragma unroll
            for (int r = 0; r < 8; ++r) {
                int m = r + 8 * gg;
                u.g.hidA[(chk * 32 + lbase + m) * 16 + e2] =
                    (_Float16)silu_f(acc[r] + bias);
            }
        }
    }
    __syncthreads();

    // ---- phase 3: GEMM2  (16 x 432) = hid(16x128) @ W2 + b2 -> wbuf(f16) ----
    // A fragments (4 x 32B per lane) cached in registers across all 7 N tiles.
    {
        v16h aF[4];
#pragma unroll
        for (int kc = 0; kc < 4; ++kc)
            aF[kc] = *(const v16h*)&u.g.hidA[(kc * 32 + lane) * 16];
        const v16h* Bp = (const v16h*)W2sw;
        for (int t = 0; t < 7; ++t) {
            int nT = wv * 7 + t;
            if (nT >= 27) break;            // wave-uniform
            v8f acc = {};
#pragma unroll
            for (int kc = 0; kc < 4; ++kc) {
                v16h b = Bp[(nT * 4 + kc) * 32 + lane];
                acc = __builtin_amdgcn_wmma_f32_16x16x32_f16(
                        false, aF[kc], false, b, (short)0, acc, false, false);
            }
            int n = nT * 16 + nlow;
            float bias = b2[n];
#pragma unroll
            for (int r = 0; r < 8; ++r) {
                int m = r + 8 * gg;
                wbuf[m * WDIM + n] = (_Float16)(acc[r] + bias);
            }
        }
    }
    __syncthreads();

    // ---- phase 4a: per-edge intermediates (reuse WMMA staging LDS) ----
    for (int idx = tid; idx < TE * V_DIM; idx += 128) {
        int m = idx >> 5, v = idx & 31;
        const float* hr  = &sHsrc[m * D_DIM];
        const float* y1  = &sY1[m * 3];
        const float* h1o = &hr[S_DIM + v * 3];
        u.p.d1[idx] = h1o[0] * y1[0] + h1o[1] * y1[1] + h1o[2] * y1[2];
        float a = 0.f;
        for (int s = 0; s < S_DIM; ++s) a += hr[s] * Wsv[s * V_DIM + v];
        sHsv[idx] = a;
        const float* By = &sBy[m * 15];
#pragma unroll
        for (int k = 0; k < 5; ++k)
            u.p.t112[idx * 5 + k] =
                h1o[0] * By[k * 3] + h1o[1] * By[k * 3 + 1] + h1o[2] * By[k * 3 + 2];
    }
    for (int idx = tid; idx < TE * L2_DIM; idx += 128) {
        int m = idx >> 4, c = idx & 15;
        const float* hr = &sHsrc[m * D_DIM];
        const float* h2 = &hr[S_DIM + 6 * V_DIM + c * 5];
        const float* y2 = &sY2[m * 5];
        float d2v = 0.f;
#pragma unroll
        for (int k = 0; k < 5; ++k) d2v += h2[k] * y2[k];
        sD2[idx] = d2v;
        float a = 0.f;
        for (int s = 0; s < S_DIM; ++s) a += hr[s] * Wsl[s * L2_DIM + c];
        sHsl[idx] = a;
        const float* By = &sBy[m * 15];
#pragma unroll
        for (int i = 0; i < 3; ++i) {
            float q = 0.f;
#pragma unroll
            for (int k = 0; k < 5; ++k) q += h2[k] * By[k * 3 + i];
            sQ[idx * 3 + i] = q;
        }
    }
    __syncthreads();

    // ---- phase 4b: assemble messages, atomic scatter-add into agg ----
    for (int idx = tid; idx < TE * MSG_DIM; idx += 128) {
        int m = idx / MSG_DIM, ch = idx - m * MSG_DIM;
        if (e0 + m >= E) continue;
        const _Float16* wr = &wbuf[m * WDIM];
        const float* hr = &sHsrc[m * D_DIM];
        const float* y1 = &sY1[m * 3];
        float val;
        if (ch < 64) {                               // m0
            int s = ch;
            float t1 = 0.f, t2 = 0.f;
            for (int v = 0; v < V_DIM; ++v)  t1 += u.p.d1[m * V_DIM + v] * Wvs[v * S_DIM + s];
            for (int c = 0; c < L2_DIM; ++c) t2 += sD2[m * L2_DIM + c] * Wls[c * S_DIM + s];
            val = (float)wr[s] * hr[s] + (float)wr[64 + s] * t1 + (float)wr[128 + s] * t2;
        } else if (ch < 160) {                       // m1o (v,i)
            int r = ch - 64; int v = r / 3, i = r - v * 3;
            int i1 = (i + 1) % 3, i2 = (i + 2) % 3;
            const float* h1o = &hr[S_DIM + v * 3];
            const float* h1e = &hr[S_DIM + 3 * V_DIM + v * 3];
            float cr = h1e[i1] * y1[i2] - h1e[i2] * y1[i1];
            float t = 0.f;
            for (int c = 0; c < L2_DIM; ++c)
                t += sQ[(m * L2_DIM + c) * 3 + i] * Wlv[c * V_DIM + v];
            val = (float)wr[192 + v] * h1o[i]
                + (float)wr[224 + v] * sHsv[m * V_DIM + v] * y1[i]
                + (float)wr[256 + v] * cr
                + (float)wr[288 + v] * t;
        } else if (ch < 256) {                       // m1e (v,i)
            int r = ch - 160; int v = r / 3, i = r - v * 3;
            int i1 = (i + 1) % 3, i2 = (i + 2) % 3;
            const float* h1o = &hr[S_DIM + v * 3];
            const float* h1e = &hr[S_DIM + 3 * V_DIM + v * 3];
            float cr = h1o[i1] * y1[i2] - h1o[i2] * y1[i1];
            val = (float)wr[320 + v] * h1e[i] + (float)wr[352 + v] * cr;
        } else {                                     // m2 (c,k)
            int r = ch - 256; int c = r / 5, k = r - c * 5;
            const float* h2 = &hr[S_DIM + 6 * V_DIM + c * 5];
            float t = 0.f;
            for (int v = 0; v < V_DIM; ++v)
                t += u.p.t112[(m * V_DIM + v) * 5 + k] * Wvl[v * L2_DIM + c];
            val = (float)wr[384 + c] * h2[k]
                + (float)wr[400 + c] * t
                + (float)wr[416 + c] * sHsl[m * L2_DIM + c] * sY2[m * 5 + k];
        }
        atomicAdd(&agg[(long)sDst[m] * MSG_DIM + ch], val);
    }
}

// ---------------------------------------------------------------------------
// Node kernel: projections + residual + FiLM + layernorm / vector RMS norms
// ---------------------------------------------------------------------------
__global__ __launch_bounds__(128)
void node_kernel(const float* __restrict__ h, const float* __restrict__ agg,
                 const float* __restrict__ temb,
                 const float* __restrict__ P0, const float* __restrict__ P1o,
                 const float* __restrict__ P1e, const float* __restrict__ P2,
                 const float* __restrict__ filmW, const float* __restrict__ filmb,
                 float* __restrict__ out)
{
    __shared__ float sAgg[D_DIM], sH[D_DIM], sT[T_DIM], sFilm[208];
    __shared__ float sRedA[128], sRedB[128];
    const int n = blockIdx.x, tid = threadIdx.x;

    for (int i = tid; i < D_DIM; i += 128) {
        sAgg[i] = agg[(long)n * D_DIM + i];
        sH[i]   = h[(long)n * D_DIM + i];
    }
    if (tid < T_DIM) sT[tid] = temb[(long)n * T_DIM + tid];
    __syncthreads();

    for (int f = tid; f < 208; f += 128) {           // FILM = 2S + 2V + L2 = 208
        float a = filmb[f];
        for (int t = 0; t < T_DIM; ++t) a += sT[t] * filmW[t * 208 + f];
        sFilm[f] = a;
    }

    // scalar channel: a0 = silu(agg0 @ P0); residual; layernorm; FiLM
    float x0 = 0.f;
    if (tid < 64) {
        float a = 0.f;
        for (int s = 0; s < 64; ++s) a += sAgg[s] * P0[s * 64 + tid];
        x0 = sH[tid] + silu_f(a);
    }
    sRedA[tid] = (tid < 64) ? x0 : 0.f;
    sRedB[tid] = (tid < 64) ? x0 * x0 : 0.f;
    for (int st = 64; st > 0; st >>= 1) {
        __syncthreads();
        if (tid < st) { sRedA[tid] += sRedA[tid + st]; sRedB[tid] += sRedB[tid + st]; }
    }
    __syncthreads();
    {
        float mu  = sRedA[0] * (1.f / 64.f);
        float var = sRedB[0] * (1.f / 64.f) - mu * mu;
        if (tid < 64)
            out[(long)n * D_DIM + tid] =
                (x0 - mu) * rsqrtf(var + 1e-6f) * (1.f + sFilm[tid]) + sFilm[64 + tid];
    }

    // odd vectors
    __syncthreads();
    float x1 = 0.f;
    if (tid < 96) {
        int w = tid / 3, i = tid - w * 3;
        float a = 0.f;
        for (int v = 0; v < 32; ++v) a += sAgg[64 + v * 3 + i] * P1o[v * 32 + w];
        x1 = sH[64 + tid] + a;
    }
    sRedA[tid] = (tid < 96) ? x1 * x1 : 0.f;
    for (int st = 64; st > 0; st >>= 1) {
        __syncthreads();
        if (tid < st) sRedA[tid] += sRedA[tid + st];
    }
    __syncthreads();
    {
        float rms = sqrtf(sRedA[0] * (1.f / 32.f) + 1e-6f);
        if (tid < 96)
            out[(long)n * D_DIM + 64 + tid] = x1 / rms * (1.f + sFilm[128 + tid / 3]);
    }

    // even vectors
    __syncthreads();
    float xe = 0.f;
    if (tid < 96) {
        int w = tid / 3, i = tid - w * 3;
        float a = 0.f;
        for (int v = 0; v < 32; ++v) a += sAgg[160 + v * 3 + i] * P1e[v * 32 + w];
        xe = sH[160 + tid] + a;
    }
    sRedA[tid] = (tid < 96) ? xe * xe : 0.f;
    for (int st = 64; st > 0; st >>= 1) {
        __syncthreads();
        if (tid < st) sRedA[tid] += sRedA[tid + st];
    }
    __syncthreads();
    {
        float rms = sqrtf(sRedA[0] * (1.f / 32.f) + 1e-6f);
        if (tid < 96)
            out[(long)n * D_DIM + 160 + tid] = xe / rms * (1.f + sFilm[160 + tid / 3]);
    }

    // l=2 channel
    __syncthreads();
    float x2 = 0.f;
    if (tid < 80) {
        int d = tid / 5, k = tid - d * 5;
        float a = 0.f;
        for (int c = 0; c < 16; ++c) a += sAgg[256 + c * 5 + k] * P2[c * 16 + d];
        x2 = sH[256 + tid] + a;
    }
    sRedA[tid] = (tid < 80) ? x2 * x2 : 0.f;
    for (int st = 64; st > 0; st >>= 1) {
        __syncthreads();
        if (tid < st) sRedA[tid] += sRedA[tid + st];
    }
    __syncthreads();
    {
        float rms = sqrtf(sRedA[0] * (1.f / 16.f) + 1e-6f);
        if (tid < 80)
            out[(long)n * D_DIM + 256 + tid] = x2 / rms * (1.f + sFilm[192 + tid / 5]);
    }
}

// ---------------------------------------------------------------------------
extern "C" void kernel_launch(void* const* d_in, const int* in_sizes, int n_in,
                              void* d_out, int out_size, void* d_ws, size_t ws_size,
                              hipStream_t stream)
{
    const float* h      = (const float*)d_in[0];
    const float* coords = (const float*)d_in[1];
    const int*   ei     = (const int*)d_in[2];
    const int*   etype  = (const int*)d_in[3];
    const int*   ebt    = (const int*)d_in[4];
    const int*   ebc    = (const int*)d_in[5];
    const int*   ebr    = (const int*)d_in[6];
    const int*   ebs    = (const int*)d_in[7];
    const float* erd    = (const float*)d_in[8];
    const float* temb   = (const float*)d_in[9];
    const float* etype_emb = (const float*)d_in[10];
    const float* bte    = (const float*)d_in[11];
    const float* bce    = (const float*)d_in[12];
    const float* bre    = (const float*)d_in[13];
    const float* bse    = (const float*)d_in[14];
    const float* refW   = (const float*)d_in[15];
    const float* refb   = (const float*)d_in[16];
    const float* W1     = (const float*)d_in[17];
    const float* b1     = (const float*)d_in[18];
    const float* W2     = (const float*)d_in[19];
    const float* b2     = (const float*)d_in[20];
    const float* Wvs    = (const float*)d_in[21];
    const float* Wls    = (const float*)d_in[22];
    const float* Wsv    = (const float*)d_in[23];
    const float* Wlv    = (const float*)d_in[24];
    const float* Wvl    = (const float*)d_in[25];
    const float* Wsl    = (const float*)d_in[26];
    const float* P0     = (const float*)d_in[27];
    const float* P1o    = (const float*)d_in[28];
    const float* P1e    = (const float*)d_in[29];
    const float* P2     = (const float*)d_in[30];
    const float* filmW  = (const float*)d_in[31];
    const float* filmb  = (const float*)d_in[32];

    const int N = in_sizes[0] / D_DIM;
    const int E = in_sizes[3];

    // workspace layout
    float* agg = (float*)d_ws;
    size_t aggBytes = (size_t)N * MSG_DIM * sizeof(float);
    size_t off = (aggBytes + 255) & ~(size_t)255;
    _Float16* W1sw = (_Float16*)((char*)d_ws + off);
    _Float16* W2sw = (_Float16*)((char*)d_ws + off + 8 * 8 * 512 * sizeof(_Float16));

    prep_kernel<<<2048, 256, 0, stream>>>(W1, W2, W1sw, W2sw, agg, (long)N * MSG_DIM);

    edge_kernel<<<(E + TE - 1) / TE, 128, 0, stream>>>(
        h, coords, ei, etype, ebt, ebc, ebr, ebs, erd, temb,
        etype_emb, bte, bce, bre, bse, refW, refb, b1, b2,
        Wvs, Wls, Wsv, Wlv, Wvl, Wsl, W1sw, W2sw, agg, E);

    node_kernel<<<N, 128, 0, stream>>>(
        h, agg, temb, P0, P1o, P1e, P2, filmW, filmb, (float*)d_out);
}
